// SeparableFCTP_6468220748598
// MI455X (gfx1250) — compile-verified, matrix operand order checked by reading, not run on hardware
//
#include <hip/hip_runtime.h>
#include <hip/hip_bf16.h>
#include <math.h>

#define E_TOTAL 100000
#define INV_SQRT3 0.5773502691896258f
#define INV_SQRT2 0.7071067811865476f
#define EPS 1e-5f

typedef __bf16 bf16_t;
typedef __attribute__((ext_vector_type(16))) __bf16 v16bf;
typedef __attribute__((ext_vector_type(8)))  float  v8f;

// ---------------- workspace (d_ws) layout: bf16 packed B fragments ----------
// Each 16(N)x32(K) B tile is stored as 32 lanes x 16 bf16 (512 bf16 = 1KB),
// matching the per-lane WMMA operand fragment, so a lane fetches its whole
// fragment with one contiguous 32B read.
#define W1P_OFF   0
#define W1P_TILES 8            // K=64 (2 kt) x N=64  (4 nt)
#define W2P_OFF   (W1P_OFF + W1P_TILES*512)
#define W2P_TILES 112          // K=64 (2 kt) x N=896 (56 nt)
#define WSP_OFF   (W2P_OFF + W2P_TILES*512)
#define WSP_TILES 192          // K=384 (12 kt) x N=256 (16 nt)
#define WVP_OFF   (WSP_OFF + WSP_TILES*512)
#define WVP_TILES 128          // K=512 (16 kt) x N=128 (8 nt)

// ---------------- LDS layout (bytes) ----------------------------------------
#define OFF_ESA   0            // edge_scalars A-frags: 2 tiles * 1KB
#define OFF_HF    2048         // h fp32 16x64
#define OFF_HA    6144         // h bf16 A-frags: 2 tiles
#define OFF_WT    8192         // w fp32 16x896 = 57344 B
#define OFF_SF    8192         //   alias: s fp32 16x256 = 16384 B
#define OFF_VF    24576        //   alias: v fp32 16x384 = 24576 B (ends 49152)
#define OFF_SCALA 65536        // scalars A-frags: 12 tiles
#define OFF_VECA  77824        // vectors A-frags: 3*16 tiles (ends 126976)
#define SMEM_BYTES 126976

// Position of element (row, kin) inside a packed 16x32 bf16 A/B fragment tile
// (in bf16 units). Derived from the CDNA5 ISA 16-bit A-matrix 16x32 layout.
__device__ __forceinline__ int fragU16Index(int row, int kin) {
  int l = row + (((kin >> 3) & 1) << 4);           // lane
  int v = ((kin >> 4) << 2) | ((kin & 7) >> 1);    // vgpr
  return l * 16 + v * 2 + (kin & 1);
}

// Repack a row-major [K][N] fp32 weight into bf16 B-fragment tiles.
__global__ void pack_b_kernel(const float* __restrict__ W, int K, int N,
                              bf16_t* __restrict__ out) {
  int gid = blockIdx.x * blockDim.x + threadIdx.x;
  int ntN = N >> 4;
  int ntiles = (K >> 5) * ntN;
  if (gid >= ntiles * 32) return;
  int lane = gid & 31;
  int tile = gid >> 5;
  int kt = tile / ntN;
  int nt = tile - kt * ntN;
  int n  = nt * 16 + (lane & 15);
  int hh = lane >> 4;
  bf16_t* dst = out + tile * 512 + lane * 16;
#pragma unroll
  for (int v = 0; v < 8; ++v) {
    int k = kt * 32 + ((v >> 2) << 4) + (hh << 3) + ((v & 3) << 1);
    dst[v * 2 + 0] = (bf16_t)W[(size_t)(k + 0) * N + n];
    dst[v * 2 + 1] = (bf16_t)W[(size_t)(k + 1) * N + n];
  }
}

__global__ __launch_bounds__(256)
void fused_fctp_kernel(const float* __restrict__ node_input,
                       const float* __restrict__ edge_attr,
                       const float* __restrict__ edge_scalars,
                       const float* __restrict__ rad_b1,
                       const float* __restrict__ rad_ln_g,
                       const float* __restrict__ rad_ln_b,
                       const float* __restrict__ rad_offset,
                       const float* __restrict__ lin_bs,
                       const float* __restrict__ ln_gs,
                       const float* __restrict__ ln_bs2,
                       const float* __restrict__ ln_gv,
                       const bf16_t* __restrict__ wpack,
                       float* __restrict__ out) {
  extern __shared__ char smem[];
  const int tid  = threadIdx.x;
  const int lane = tid & 31;
  const int wave = tid >> 5;
  const int e0   = blockIdx.x * 16;

  // ---- P0: load edge_scalars tile (16x64) -> bf16 A fragments ----
  {
    bf16_t* esA = (bf16_t*)(smem + OFF_ESA);
    int row = tid >> 4, j = tid & 15;
    const float* src = edge_scalars + (size_t)(e0 + row) * 64;
#pragma unroll
    for (int i = 0; i < 4; ++i) {
      int k = j * 4 + i;
      esA[(k >> 5) * 512 + fragU16Index(row, k & 31)] = (bf16_t)src[k];
    }
  }
  __syncthreads();

  // ---- P1: h = es @ rad_W1 + b1   (M=16, N=64, K=64) ----
  if (wave < 4) {
    const bf16_t* esA = (const bf16_t*)(smem + OFF_ESA);
    float* hF = (float*)(smem + OFF_HF);
    int nt = wave;
    v8f acc = {};
#pragma unroll
    for (int kt = 0; kt < 2; ++kt) {
      v16bf a = *(const v16bf*)(esA + kt * 512 + lane * 16);
      v16bf b = *(const v16bf*)(wpack + W1P_OFF + (kt * 4 + nt) * 512 + lane * 16);
      acc = __builtin_amdgcn_wmma_f32_16x16x32_bf16(false, a, false, b,
                                                    (short)0, acc, false, false);
    }
    int n = nt * 16 + (lane & 15);
    float bias = rad_b1[n];
    int hh = lane >> 4;
#pragma unroll
    for (int r = 0; r < 8; ++r) hF[(r + (hh << 3)) * 64 + n] = acc[r] + bias;
  }
  __syncthreads();

  // ---- P2: LayerNorm + SiLU over 64 features, repack bf16 A frags ----
  {
    const float* hF = (const float*)(smem + OFF_HF);
    bf16_t* hA = (bf16_t*)(smem + OFF_HA);
    int row = tid >> 4, j = tid & 15;
    float x[4], sum = 0.f, sq = 0.f;
#pragma unroll
    for (int i = 0; i < 4; ++i) {
      x[i] = hF[row * 64 + j * 4 + i];
      sum += x[i]; sq += x[i] * x[i];
    }
#pragma unroll
    for (int m = 1; m < 16; m <<= 1) {
      sum += __shfl_xor(sum, m, 32);
      sq  += __shfl_xor(sq,  m, 32);
    }
    float mu   = sum * (1.f / 64.f);
    float var  = sq * (1.f / 64.f) - mu * mu;
    float rstd = rsqrtf(var + EPS);
#pragma unroll
    for (int i = 0; i < 4; ++i) {
      int k = j * 4 + i;
      float hv = (x[i] - mu) * rstd * rad_ln_g[k] + rad_ln_b[k];
      hv = hv / (1.f + __expf(-hv));                      // SiLU
      hA[(k >> 5) * 512 + fragU16Index(row, k & 31)] = (bf16_t)hv;
    }
  }
  __syncthreads();

  // ---- P3: w = h @ rad_W2 + offset   (M=16, N=896, K=64), fp32 in LDS ----
  {
    const bf16_t* hA = (const bf16_t*)(smem + OFF_HA);
    float* wT = (float*)(smem + OFF_WT);
    int nn = lane & 15, hh = lane >> 4;
    for (int i = 0; i < 7; ++i) {
      int nt = wave * 7 + i;
      v8f acc = {};
#pragma unroll
      for (int kt = 0; kt < 2; ++kt) {
        v16bf a = *(const v16bf*)(hA + kt * 512 + lane * 16);
        v16bf b = *(const v16bf*)(wpack + W2P_OFF + (kt * 56 + nt) * 512 + lane * 16);
        acc = __builtin_amdgcn_wmma_f32_16x16x32_bf16(false, a, false, b,
                                                      (short)0, acc, false, false);
      }
      int n = nt * 16 + nn;
      float off = rad_offset[n];
#pragma unroll
      for (int r = 0; r < 8; ++r) wT[(r + (hh << 3)) * 896 + n] = acc[r] + off;
    }
  }
  __syncthreads();

  // ---- P4: tensor product -> bf16 A frags for scalar (K=384) / vector (K=512 x3) GEMMs ----
  {
    const float* wT = (const float*)(smem + OFF_WT);
    bf16_t* scalA = (bf16_t*)(smem + OFF_SCALA);
    bf16_t* vecA  = (bf16_t*)(smem + OFF_VECA);
    int row = tid >> 4, j = tid & 15;
    int e = e0 + row;
    const float* ni = node_input + (size_t)e * 640;
    float y0  = edge_attr[e * 4 + 0];
    float y1x = edge_attr[e * 4 + 1];
    float y1y = edge_attr[e * 4 + 2];
    float y1z = edge_attr[e * 4 + 3];
    const float* wrow = wT + row * 896;

    for (int c = j; c < 384; c += 16) {           // scalar channels
      float val;
      if (c < 256) {
        val = ni[c] * y0 * wrow[c];                               // x0*y0*wA
      } else {
        int u = c - 256;
        float a0 = ni[256 + u * 3 + 0], a1 = ni[256 + u * 3 + 1], a2 = ni[256 + u * 3 + 2];
        val = INV_SQRT3 * (a0 * y1x + a1 * y1y + a2 * y1z) * wrow[640 + u];  // dot*wD
      }
      scalA[(c >> 5) * 512 + fragU16Index(row, c & 31)] = (bf16_t)val;
    }
    for (int c = j; c < 512; c += 16) {           // vector channels (3 comps)
      float v0, v1, v2;
      if (c < 256) {
        float base = ni[c] * wrow[256 + c];                       // x0*wB * y1
        v0 = base * y1x; v1 = base * y1y; v2 = base * y1z;
      } else if (c < 384) {
        int u = c - 256;
        float coef = y0 * wrow[512 + u];                          // x1*y0*wC
        v0 = ni[256 + u * 3 + 0] * coef;
        v1 = ni[256 + u * 3 + 1] * coef;
        v2 = ni[256 + u * 3 + 2] * coef;
      } else {
        int u = c - 384;
        float a0 = ni[256 + u * 3 + 0], a1 = ni[256 + u * 3 + 1], a2 = ni[256 + u * 3 + 2];
        float coef = INV_SQRT2 * wrow[768 + u];                   // cross*wE
        v0 = (a1 * y1z - a2 * y1y) * coef;
        v1 = (a2 * y1x - a0 * y1z) * coef;
        v2 = (a0 * y1y - a1 * y1x) * coef;
      }
      int fi = (c >> 5) * 512 + fragU16Index(row, c & 31);
      vecA[(0 * 16) * 512 + fi] = (bf16_t)v0;
      vecA[(1 * 16) * 512 + fi] = (bf16_t)v1;
      vecA[(2 * 16) * 512 + fi] = (bf16_t)v2;
    }
  }
  __syncthreads();

  // ---- P5: s = scalars @ lin_Ws + bs   (M=16, N=256, K=384) ----
  {
    const bf16_t* scalA = (const bf16_t*)(smem + OFF_SCALA);
    float* sF = (float*)(smem + OFF_SF);
    int nn = lane & 15, hh = lane >> 4;
#pragma unroll
    for (int i = 0; i < 2; ++i) {
      int nt = wave + i * 8;
      v8f acc = {};
      for (int kt = 0; kt < 12; ++kt) {
        v16bf a = *(const v16bf*)(scalA + kt * 512 + lane * 16);
        v16bf b = *(const v16bf*)(wpack + WSP_OFF + (kt * 16 + nt) * 512 + lane * 16);
        acc = __builtin_amdgcn_wmma_f32_16x16x32_bf16(false, a, false, b,
                                                      (short)0, acc, false, false);
      }
      int n = nt * 16 + nn;
      float bias = lin_bs[n];
#pragma unroll
      for (int r = 0; r < 8; ++r) sF[(r + (hh << 3)) * 256 + n] = acc[r] + bias;
    }
  }

  // ---- P6: v[:, :, m] = vectors_m @ lin_Wv   (3 x (M=16, N=128, K=512)) ----
  {
    const bf16_t* vecA = (const bf16_t*)(smem + OFF_VECA);
    float* vF = (float*)(smem + OFF_VF);
    int nn = lane & 15, hh = lane >> 4;
#pragma unroll
    for (int i = 0; i < 3; ++i) {
      int job = wave + i * 8;
      int m = job >> 3, nt = job & 7;
      v8f acc = {};
      for (int kt = 0; kt < 16; ++kt) {
        v16bf a = *(const v16bf*)(vecA + (m * 16 + kt) * 512 + lane * 16);
        v16bf b = *(const v16bf*)(wpack + WVP_OFF + (kt * 8 + nt) * 512 + lane * 16);
        acc = __builtin_amdgcn_wmma_f32_16x16x32_bf16(false, a, false, b,
                                                      (short)0, acc, false, false);
      }
      int n = nt * 16 + nn;
#pragma unroll
      for (int r = 0; r < 8; ++r)
        vF[(r + (hh << 3)) * 384 + m * 128 + n] = acc[r];
    }
  }
  __syncthreads();

  // ---- P7: scalar LayerNorm + vector RMS norm, write 640 fp32 / edge ----
  {
    const float* sF = (const float*)(smem + OFF_SF);
    const float* vF = (const float*)(smem + OFF_VF);
    int row = tid >> 4, j = tid & 15;
    int e = e0 + row;
    float sum = 0.f, sq = 0.f, vs = 0.f;
    for (int c = j; c < 256; c += 16) {
      float x = sF[row * 256 + c];
      sum += x; sq += x * x;
    }
    for (int idx = j; idx < 384; idx += 16) {
      float x = vF[row * 384 + idx];
      vs += x * x;
    }
#pragma unroll
    for (int m = 1; m < 16; m <<= 1) {
      sum += __shfl_xor(sum, m, 32);
      sq  += __shfl_xor(sq,  m, 32);
      vs  += __shfl_xor(vs,  m, 32);
    }
    float mu   = sum * (1.f / 256.f);
    float var  = sq * (1.f / 256.f) - mu * mu;
    float rstd = rsqrtf(var + EPS);
    float rv   = rsqrtf(vs * (1.f / 384.f) + EPS);
    float* orow = out + (size_t)e * 640;
    for (int c = j; c < 256; c += 16)
      orow[c] = (sF[row * 256 + c] - mu) * rstd * ln_gs[c] + ln_bs2[c];
    for (int idx = j; idx < 384; idx += 16) {
      int m = idx >> 7, u = idx & 127;
      orow[256 + u * 3 + m] = vF[row * 384 + idx] * rv * ln_gv[u];
    }
  }
}

extern "C" void kernel_launch(void* const* d_in, const int* in_sizes, int n_in,
                              void* d_out, int out_size, void* d_ws, size_t ws_size,
                              hipStream_t stream) {
  const float* node_input   = (const float*)d_in[0];
  const float* edge_attr    = (const float*)d_in[1];
  const float* edge_scalars = (const float*)d_in[2];
  const float* rad_W1       = (const float*)d_in[3];
  const float* rad_b1       = (const float*)d_in[4];
  const float* rad_ln_g     = (const float*)d_in[5];
  const float* rad_ln_b     = (const float*)d_in[6];
  const float* rad_W2       = (const float*)d_in[7];
  const float* rad_offset   = (const float*)d_in[8];
  const float* lin_Ws       = (const float*)d_in[9];
  const float* lin_bs       = (const float*)d_in[10];
  const float* lin_Wv       = (const float*)d_in[11];
  const float* ln_gs        = (const float*)d_in[12];
  const float* ln_bs2       = (const float*)d_in[13];
  const float* ln_gv        = (const float*)d_in[14];
  bf16_t* wpack = (bf16_t*)d_ws;

  // One-time (per launch) weight repack into bf16 B fragments (L2-resident).
  pack_b_kernel<<<(W1P_TILES * 32 + 255) / 256, 256, 0, stream>>>(rad_W1, 64, 64,  wpack + W1P_OFF);
  pack_b_kernel<<<(W2P_TILES * 32 + 255) / 256, 256, 0, stream>>>(rad_W2, 64, 896, wpack + W2P_OFF);
  pack_b_kernel<<<(WSP_TILES * 32 + 255) / 256, 256, 0, stream>>>(lin_Ws, 384, 256, wpack + WSP_OFF);
  pack_b_kernel<<<(WVP_TILES * 32 + 255) / 256, 256, 0, stream>>>(lin_Wv, 512, 128, wpack + WVP_OFF);

  fused_fctp_kernel<<<E_TOTAL / 16, 256, SMEM_BYTES, stream>>>(
      node_input, edge_attr, edge_scalars, rad_b1, rad_ln_g, rad_ln_b,
      rad_offset, lin_bs, ln_gs, ln_bs2, ln_gv, wpack, (float*)d_out);
}
